// DecoderLayer_57123065036932
// MI455X (gfx1250) — matrix operand order repaired and checked by reading, not compiled, not run on hardware
//
#include <hip/hip_runtime.h>
#include <hip/hip_bf16.h>
#include <stdint.h>

// ---------------------------------------------------------------------------
// CDNA5 / gfx1250 transformer decoder layer.
// bf16 WMMA for all matmuls; TDM (tensor_load_to_lds) + register-pipelined
// staging for the GEMMs; flash attention with online softmax.
// ---------------------------------------------------------------------------

typedef __attribute__((ext_vector_type(16))) __bf16 v16bf;
typedef __attribute__((ext_vector_type(8)))  float  v8f;
typedef __attribute__((ext_vector_type(4)))  unsigned int u32x4;
typedef __attribute__((ext_vector_type(8)))  int    i32x8;
typedef __attribute__((ext_vector_type(4)))  int    i32x4;
typedef unsigned short ushort_t;
typedef unsigned int   uint_t;

union Frag16 { uint_t u[8]; v16bf v; };

#if __has_builtin(__builtin_amdgcn_tensor_load_to_lds)
#define USE_TDM 1
#else
#define USE_TDM 0
#endif

#if __has_builtin(__builtin_amdgcn_s_wait_tensorcnt)
#define TDM_WAIT() __builtin_amdgcn_s_wait_tensorcnt(0)
#else
#define TDM_WAIT() asm volatile("s_wait_tensorcnt 0x0" ::: "memory")
#endif

__device__ __forceinline__ ushort_t f2bf(float f) {
  uint_t u = __float_as_uint(f);
  u += 0x7FFFu + ((u >> 16) & 1u);          // round-to-nearest-even
  return (ushort_t)(u >> 16);
}

constexpr int Bn = 4, Tn = 2048, Hn = 16, DHn = 64, Dn = 1024, DFFn = 4096;

#if USE_TDM
// ---------------------------------------------------------------------------
// TDM: DMA a 2-D tile (128 rows x 64 bytes) of bf16 A into LDS, inserting
// 16 bytes of padding after every 64 bytes (pad_interval = 16 DWORDs,
// pad_amount = 4 DWORDs) so rows land at the 80-byte LDS stride the WMMA
// fragment gathers expect. D# packed per CDNA5 ISA 8.3/8.4.
// ---------------------------------------------------------------------------
__device__ __forceinline__ void tdm_load_a_tile(const ushort_t* gsrc,
                                                unsigned ldsByteOff,
                                                unsigned strideBytes) {
  unsigned long long ga = (unsigned long long)(size_t)gsrc;
  u32x4 g0;
  g0[0] = 1u;                                            // count=1, user mode
  g0[1] = ldsByteOff;                                    // lds_addr
  g0[2] = (unsigned)(ga & 0xFFFFFFFFull);                // global_addr[31:0]
  g0[3] = (unsigned)((ga >> 32) & 0x01FFFFFFull)         // global_addr[56:32]
        | (2u << 30);                                    // type = 2 (image)
  i32x8 g1;
  const unsigned tileW = 64u;                            // tile_dim0 (bytes)
  const unsigned tileH = 128u;                           // tile_dim1 (rows)
  g1[0] = (int)((1u << 20) | (3u << 22) | (3u << 25));   // pad_en, 16dw, 4dw
  g1[1] = (int)((strideBytes & 0xFFFFu) << 16);          // tensor_dim0 lo16
  g1[2] = (int)((strideBytes >> 16) | (tileH << 16));    // td0 hi | td1 lo16
  g1[3] = (int)(tileW << 16);                            // td1 hi | tile_dim0
  g1[4] = (int)tileH;                                    // tile_dim1 (dim2=0)
  g1[5] = (int)strideBytes;                              // td0_stride lo32
  g1[6] = 0;                                             // td0_stride hi16
  g1[7] = 0;
  i32x4 z4 = {0, 0, 0, 0};
#if defined(__clang_major__) && __clang_major__ >= 23
  i32x8 z8 = {0, 0, 0, 0, 0, 0, 0, 0};
  __builtin_amdgcn_tensor_load_to_lds(g0, g1, z4, z4, z8, 0);
#else
  __builtin_amdgcn_tensor_load_to_lds(g0, g1, z4, z4, 0);
#endif
}
#endif  // USE_TDM

// ---------------------------------------------------------------------------
// f32 -> bf16 bulk convert
// ---------------------------------------------------------------------------
__global__ __launch_bounds__(256) void k_f32_to_bf16(
    const float* __restrict__ in, ushort_t* __restrict__ out, size_t n) {
  size_t i = (size_t)blockIdx.x * blockDim.x + threadIdx.x;
  size_t stride = (size_t)gridDim.x * blockDim.x;
  for (; i < n; i += stride) out[i] = f2bf(in[i]);
}

// ---------------------------------------------------------------------------
// bf16 GEMM: C[M,N] = A[M,K] * B[K,N] + bias.
// Block: 256 threads (8 waves), tile 128x128, K-step 32.
// A tiles: TDM async DMA into double-buffered LDS (overlaps WMMA compute,
// tracked by TENSORcnt).  B tiles: global->VGPR issued one k-block ahead
// (tracked by LOADcnt), transposed into LDS between barriers.
// MODE bits: 1 = f32 out, 2 = bf16 out, 4 = relu.
// ---------------------------------------------------------------------------
template <int MODE>
__global__ __launch_bounds__(256) void k_gemm_bf16(
    const ushort_t* __restrict__ A, const ushort_t* __restrict__ Bw,
    const float* __restrict__ bias, float* __restrict__ outF,
    ushort_t* __restrict__ outB, int M, int N, int K) {
  __shared__ __align__(16) ushort_t ldsA[2][128 * 40];  // [row][k], stride 40
  __shared__ __align__(16) ushort_t ldsBT[128 * 40];    // transposed [n][k]

  const int m0 = blockIdx.y * 128;
  const int n0 = blockIdx.x * 128;
  const int wave = threadIdx.x >> 5;
  const int lane = threadIdx.x & 31;
  const int wm = wave >> 1, wn = wave & 1;
  const int lh = lane >> 4, ll = lane & 15;
  const int nK = K >> 5;

  v8f acc[2][4];
  for (int a = 0; a < 2; a++)
    for (int b = 0; b < 4; b++)
      for (int v = 0; v < 8; v++) acc[a][b][v] = 0.0f;

  auto stageA = [&](int buf, int k0) {
#if USE_TDM
    if (wave == 0)
      tdm_load_a_tile(A + (size_t)m0 * K + k0,
                      (unsigned)(size_t)&ldsA[buf][0], (unsigned)K * 2u);
#else
#pragma unroll
    for (int it = 0; it < 2; it++) {
      int i = threadIdx.x + it * 256;
      int row = i >> 2, ch = i & 3;
      uint4 d = *(const uint4*)(A + (size_t)(m0 + row) * K + k0 + ch * 8);
      *(uint4*)&ldsA[buf][row * 40 + ch * 8] = d;
    }
#endif
  };
  auto stageAwait = [&]() {
#if USE_TDM
    if (wave == 0) TDM_WAIT();
#endif
  };
  auto loadB = [&](int k0, uint4* reg) {
#pragma unroll
    for (int it = 0; it < 2; it++) {
      int i = threadIdx.x + it * 256;
      int kr = i >> 4, ch = i & 15;
      reg[it] = *(const uint4*)(Bw + (size_t)(k0 + kr) * N + n0 + ch * 8);
    }
  };
  auto storeB = [&](const uint4* reg) {
#pragma unroll
    for (int it = 0; it < 2; it++) {
      int i = threadIdx.x + it * 256;
      int kr = i >> 4, ch = i & 15;
      const ushort_t* p = (const ushort_t*)&reg[it];
#pragma unroll
      for (int e = 0; e < 8; e++) ldsBT[(ch * 8 + e) * 40 + kr] = p[e];
    }
  };

  // prologue: stage k-block 0
  uint4 breg[2];
  stageA(0, 0);
  loadB(0, breg);
  storeB(breg);
  stageAwait();
  __syncthreads();

  for (int kk = 0; kk < nK; kk++) {
    const int cur = kk & 1;
    const bool more = (kk + 1) < nK;
    if (more) {
      stageA(cur ^ 1, (kk + 1) << 5);   // TDM DMA overlaps compute below
      loadB((kk + 1) << 5, breg);       // global loads overlap compute below
    }

    Frag16 af[2], bf[4];
    const uint_t* Au = (const uint_t*)ldsA[cur];
    const uint_t* Bu = (const uint_t*)ldsBT;
#pragma unroll
    for (int rt = 0; rt < 2; rt++) {
      int row = wm * 32 + rt * 16 + ll;
#pragma unroll
      for (int j = 0; j < 8; j++) {
        int k = 2 * j + 16 * lh;  // A 16x32, contiguous k = 16*lh + s
        af[rt].u[j] = Au[(row * 40 + k) >> 1];
      }
    }
#pragma unroll
    for (int ct = 0; ct < 4; ct++) {
      int col = wn * 64 + ct * 16 + ll;
#pragma unroll
      for (int j = 0; j < 8; j++)                               // B 32x16, same contiguous k order as A
        bf[ct].u[j] = Bu[((col * 40 + lh * 16) >> 1) + j];
    }
#pragma unroll
    for (int rt = 0; rt < 2; rt++)
#pragma unroll
      for (int ct = 0; ct < 4; ct++)
        acc[rt][ct] = __builtin_amdgcn_wmma_f32_16x16x32_bf16(
            false, af[rt].v, false, bf[ct].v, (short)0, acc[rt][ct], false, false);

    __syncthreads();                    // everyone done reading ldsBT
    if (more) {
      storeB(breg);
      stageAwait();                     // TENSORcnt==0: next A tile landed
      __syncthreads();                  // next buffers ready
    }
  }

#pragma unroll
  for (int rt = 0; rt < 2; rt++)
    for (int ct = 0; ct < 4; ct++)
      for (int v = 0; v < 8; v++) {
        int r = m0 + wm * 32 + rt * 16 + v + 8 * lh;
        int c = n0 + wn * 64 + ct * 16 + ll;
        float val = acc[rt][ct][v] + bias[c];
        if (MODE & 4) val = fmaxf(val, 0.0f);
        if (MODE & 1) outF[(size_t)r * N + c] = val;
        if (MODE & 2) outB[(size_t)r * N + c] = f2bf(val);
      }
}

// ---------------------------------------------------------------------------
// Repack qkv[B*T, 3D] (col = h*192 + dh*3 + which) -> Q/K/V [B,H,T,DH] bf16
// ---------------------------------------------------------------------------
__global__ __launch_bounds__(256) void k_repack_qkv(
    const ushort_t* __restrict__ qkv, ushort_t* __restrict__ Qb,
    ushort_t* __restrict__ Kb, ushort_t* __restrict__ Vb) {
  size_t idx = (size_t)blockIdx.x * 256 + threadIdx.x;  // < B*T*3D
  int col = (int)(idx % (3 * Dn));
  size_t rowbt = idx / (3 * Dn);
  int h = col / (DHn * 3);
  int rr = col % (DHn * 3);
  int dh = rr / 3, wch = rr % 3;
  size_t o = ((rowbt / Tn * Hn + h) * (size_t)Tn + (rowbt % Tn)) * DHn + dh;
  ushort_t v = qkv[idx];
  ushort_t* dst = (wch == 0) ? Qb : (wch == 1) ? Kb : Vb;
  dst[o] = v;
}

// ---------------------------------------------------------------------------
// Flash attention: block = 128 threads (4 waves), 64 query rows per block,
// each wave owns a 16-row Q tile. Key loop in 64-key blocks via LDS.
// ---------------------------------------------------------------------------
__global__ __launch_bounds__(128) void k_attention(
    const ushort_t* __restrict__ Qb, const ushort_t* __restrict__ Kb,
    const ushort_t* __restrict__ Vb, const float* __restrict__ mask,
    float* __restrict__ y) {
  __shared__ __align__(16) ushort_t ldsK[64 * 72];      // [key][feat], stride 72
  __shared__ __align__(16) ushort_t ldsVT[64 * 72];     // [dh][key], stride 72
  __shared__ __align__(16) ushort_t ldsP[4 * 16 * 72];  // per-wave P staging

  const int bh = blockIdx.y;
  const int bI = bh >> 4, hI = bh & 15;
  const int wave = threadIdx.x >> 5;
  const int lane = threadIdx.x & 31;
  const int lh = lane >> 4, ll = lane & 15;
  const int q0 = blockIdx.x * 64 + wave * 16;

  const ushort_t* Qh = Qb + (size_t)bh * Tn * DHn;
  const ushort_t* Kh = Kb + (size_t)bh * Tn * DHn;
  const ushort_t* Vh = Vb + (size_t)bh * Tn * DHn;

  // Q A-fragments (16 rows x 64 feats = 2 k-steps), resident for whole kernel
  Frag16 qf[2];
  {
    const uint_t* Qu = (const uint_t*)Qh;
    int qm = q0 + ll;
#pragma unroll
    for (int ks = 0; ks < 2; ks++)
#pragma unroll
      for (int j = 0; j < 8; j++) {
        int k = 2 * j + 16 * lh + ks * 32;
        qf[ks].u[j] = Qu[(qm * DHn + k) >> 1];
      }
  }

  float mrow[8], lrow[8];
  v8f oacc[4];
  for (int v = 0; v < 8; v++) { mrow[v] = -1.0e30f; lrow[v] = 0.0f; }
  for (int dt = 0; dt < 4; dt++)
    for (int v = 0; v < 8; v++) oacc[dt][v] = 0.0f;

  for (int kb = 0; kb < Tn / 64; kb++) {
    {  // stage K (row major) and V (transposed) blocks: 64x64 bf16 each
      const uint4* Ksrc = (const uint4*)(Kh + (size_t)kb * 64 * DHn);
      const uint4* Vsrc = (const uint4*)(Vh + (size_t)kb * 64 * DHn);
#pragma unroll
      for (int it = 0; it < 4; it++) {
        int i = threadIdx.x + it * 128;  // 0..511
        int row = i >> 3, ch = i & 7;
        uint4 d = Ksrc[i];
        *(uint4*)&ldsK[row * 72 + ch * 8] = d;
        uint4 dv = Vsrc[i];
        const ushort_t* p = (const ushort_t*)&dv;
#pragma unroll
        for (int e = 0; e < 8; e++) ldsVT[(ch * 8 + e) * 72 + row] = p[e];
      }
    }
    __syncthreads();

    // S = (Q K^T) * scale + mask-bias : 4 col-tiles of 16 keys
    float sv[4][8];
    const uint_t* Ku = (const uint_t*)ldsK;
#pragma unroll
    for (int ct = 0; ct < 4; ct++) {
      v8f sacc;
      for (int v = 0; v < 8; v++) sacc[v] = 0.0f;
#pragma unroll
      for (int ks = 0; ks < 2; ks++) {
        Frag16 bf;
        int key = ct * 16 + ll;
        int base = lh * 16 + ks * 32;
#pragma unroll
        for (int j = 0; j < 8; j++) bf.u[j] = Ku[((key * 72 + base) >> 1) + j];
        sacc = __builtin_amdgcn_wmma_f32_16x16x32_bf16(
            false, qf[ks].v, false, bf.v, (short)0, sacc, false, false);
      }
#pragma unroll
      for (int v = 0; v < 8; v++) {
        int trow = q0 + v + 8 * lh;
        int scol = kb * 64 + ct * 16 + ll;
        float mk = mask[((size_t)bI * Tn + trow) * Tn + scol];
        sv[ct][v] = sacc[v] * 0.125f + (1.0f - mk) * -10000.0f;
      }
    }

    // online softmax (row reductions across the 16 lanes of each half-wave)
    float alpha[8];
#pragma unroll
    for (int v = 0; v < 8; v++) {
      float mx = fmaxf(fmaxf(sv[0][v], sv[1][v]), fmaxf(sv[2][v], sv[3][v]));
#pragma unroll
      for (int off = 1; off < 16; off <<= 1)
        mx = fmaxf(mx, __shfl_xor(mx, off, 32));
      float mnew = fmaxf(mrow[v], mx);
      alpha[v] = __expf(mrow[v] - mnew);
      float rs = 0.0f;
#pragma unroll
      for (int ct = 0; ct < 4; ct++) {
        float p = __expf(sv[ct][v] - mnew);
        sv[ct][v] = p;
        rs += p;
      }
#pragma unroll
      for (int off = 1; off < 16; off <<= 1) rs += __shfl_xor(rs, off, 32);
      lrow[v] = lrow[v] * alpha[v] + rs;
      mrow[v] = mnew;
    }

#pragma unroll
    for (int dt = 0; dt < 4; dt++)
      for (int v = 0; v < 8; v++) oacc[dt][v] *= alpha[v];

    // P (bf16) through per-wave LDS staging to re-layout as WMMA A fragments
    ushort_t* myP = ldsP + wave * 16 * 72;
#pragma unroll
    for (int ct = 0; ct < 4; ct++)
      for (int v = 0; v < 8; v++)
        myP[(v + 8 * lh) * 72 + ct * 16 + ll] = f2bf(sv[ct][v]);

    const uint_t* Pu = (const uint_t*)myP;
    const uint_t* Vu = (const uint_t*)ldsVT;
#pragma unroll
    for (int dt = 0; dt < 4; dt++) {
#pragma unroll
      for (int ks = 0; ks < 2; ks++) {
        Frag16 pa, vf;
#pragma unroll
        for (int j = 0; j < 8; j++) {
          int k = 2 * j + 16 * lh + ks * 32;
          pa.u[j] = Pu[(ll * 72 + k) >> 1];
        }
        int dh = dt * 16 + ll;
        int kbase = lh * 16 + ks * 32;
#pragma unroll
        for (int j = 0; j < 8; j++) vf.u[j] = Vu[((dh * 72 + kbase) >> 1) + j];
        oacc[dt] = __builtin_amdgcn_wmma_f32_16x16x32_bf16(
            false, pa.v, false, vf.v, (short)0, oacc[dt], false, false);
      }
    }
    __syncthreads();
  }

#pragma unroll
  for (int dt = 0; dt < 4; dt++)
    for (int v = 0; v < 8; v++) {
      int trow = q0 + v + 8 * lh;
      int dh = dt * 16 + ll;
      y[((size_t)bI * Tn + trow) * Dn + hI * DHn + dh] = oacc[dt][v] / lrow[v];
    }
}

// ---------------------------------------------------------------------------
// Fused residual + LayerNorm. MODE bits: 1 = f32 out, 2 = bf16 out.
// One block (256 threads) per row of D=1024.
// ---------------------------------------------------------------------------
template <int MODE>
__global__ __launch_bounds__(256) void k_resid_ln(
    const float* __restrict__ a, const float* __restrict__ r,
    const float* __restrict__ w, const float* __restrict__ bb,
    float* __restrict__ outF, ushort_t* __restrict__ outB) {
  const int row = blockIdx.x;
  const float* pa = a + (size_t)row * Dn;
  const float* pr = r + (size_t)row * Dn;
  __shared__ float redS[8], redS2[8];
  float vals[4];
  float s = 0.f, s2 = 0.f;
#pragma unroll
  for (int i = 0; i < 4; i++) {
    int c = threadIdx.x + i * 256;
    float v = pa[c] + pr[c];
    vals[i] = v;
    s += v;
    s2 += v * v;
  }
#pragma unroll
  for (int off = 1; off < 32; off <<= 1) {
    s += __shfl_xor(s, off, 32);
    s2 += __shfl_xor(s2, off, 32);
  }
  int wave = threadIdx.x >> 5, lane = threadIdx.x & 31;
  if (lane == 0) { redS[wave] = s; redS2[wave] = s2; }
  __syncthreads();
  if (wave == 0) {
    float ts = (lane < 8) ? redS[lane] : 0.f;
    float ts2 = (lane < 8) ? redS2[lane] : 0.f;
#pragma unroll
    for (int off = 1; off < 8; off <<= 1) {
      ts += __shfl_xor(ts, off, 32);
      ts2 += __shfl_xor(ts2, off, 32);
    }
    if (lane == 0) { redS[0] = ts; redS2[0] = ts2; }
  }
  __syncthreads();
  float mu = redS[0] * (1.0f / Dn);
  float var = redS2[0] * (1.0f / Dn) - mu * mu;
  float inv = rsqrtf(var + 1e-5f);
#pragma unroll
  for (int i = 0; i < 4; i++) {
    int c = threadIdx.x + i * 256;
    float o = (vals[i] - mu) * inv * w[c] + bb[c];
    if (MODE & 1) outF[(size_t)row * Dn + c] = o;
    if (MODE & 2) outB[(size_t)row * Dn + c] = f2bf(o);
  }
}

// ---------------------------------------------------------------------------
// Orchestration
// ---------------------------------------------------------------------------
extern "C" void kernel_launch(void* const* d_in, const int* in_sizes, int n_in,
                              void* d_out, int out_size, void* d_ws, size_t ws_size,
                              hipStream_t stream) {
  (void)in_sizes; (void)n_in; (void)out_size; (void)ws_size;
  const float* x     = (const float*)d_in[0];
  const float* mask  = (const float*)d_in[1];
  const float* w_qkv = (const float*)d_in[2];
  const float* b_qkv = (const float*)d_in[3];
  const float* w_ff  = (const float*)d_in[4];
  const float* b_ff  = (const float*)d_in[5];
  const float* w_out = (const float*)d_in[6];
  const float* b_out = (const float*)d_in[7];
  const float* ln1w  = (const float*)d_in[8];
  const float* ln1b  = (const float*)d_in[9];
  const float* ln2w  = (const float*)d_in[10];
  const float* ln2b  = (const float*)d_in[11];
  float* out = (float*)d_out;
  char* ws = (char*)d_ws;

  // workspace arena (lifetime-aliased)
  constexpr size_t SZ_XB    = (size_t)Bn * Tn * Dn * 2;       // 16 MB (also x1b)
  constexpr size_t SZ_WQKVB = (size_t)Dn * 3 * Dn * 2;        //  6 MB
  constexpr size_t SZ_WFFB  = (size_t)Dn * DFFn * 2;          //  8 MB
  constexpr size_t SZ_WOUTB = (size_t)DFFn * Dn * 2;          //  8 MB
  constexpr size_t SZ_QKVB  = (size_t)Bn * Tn * 3 * Dn * 2;   // 48 MB (also y2 f32)
  constexpr size_t SZ_HEAD  = (size_t)Bn * Hn * Tn * DHn * 2; // 16 MB each Q/K/V
  constexpr size_t SZ_Y     = (size_t)Bn * Tn * Dn * 4;       // 32 MB

  const size_t o_xb    = 0;
  const size_t o_wqkvb = o_xb + SZ_XB;
  const size_t o_wffb  = o_wqkvb + SZ_WQKVB;
  const size_t o_woutb = o_wffb + SZ_WFFB;
  const size_t o_qkvb  = o_woutb + SZ_WOUTB;
  const size_t o_qb    = o_qkvb + SZ_QKVB;   // QKV region reused as FFN hidden
  const size_t o_kb    = o_qb + SZ_HEAD;
  const size_t o_vb    = o_kb + SZ_HEAD;
  const size_t o_y     = o_vb + SZ_HEAD;
  const size_t o_x1    = o_y + SZ_Y;

  ushort_t* xb    = (ushort_t*)(ws + o_xb);
  ushort_t* x1b   = (ushort_t*)(ws + o_xb);     // alias: xb dead after QKV GEMM
  ushort_t* wqkvb = (ushort_t*)(ws + o_wqkvb);
  ushort_t* wffb  = (ushort_t*)(ws + o_wffb);
  ushort_t* woutb = (ushort_t*)(ws + o_woutb);
  ushort_t* qkvb  = (ushort_t*)(ws + o_qkvb);
  float*    y2    = (float*)   (ws + o_qkvb);   // alias: qkvb dead after repack
  ushort_t* Qb    = (ushort_t*)(ws + o_qb);
  ushort_t* Kb    = (ushort_t*)(ws + o_kb);
  ushort_t* Vb    = (ushort_t*)(ws + o_vb);
  ushort_t* h1b   = (ushort_t*)(ws + o_qb);     // alias: Q/K/V + y dead after LN1
  float*    yattn = (float*)   (ws + o_y);
  float*    x1    = (float*)   (ws + o_x1);

  // 1) downconvert activations + weights to bf16
  k_f32_to_bf16<<<4096, 256, 0, stream>>>(x,     xb,    (size_t)Bn * Tn * Dn);
  k_f32_to_bf16<<<4096, 256, 0, stream>>>(w_qkv, wqkvb, (size_t)Dn * 3 * Dn);
  k_f32_to_bf16<<<4096, 256, 0, stream>>>(w_ff,  wffb,  (size_t)Dn * DFFn);
  k_f32_to_bf16<<<4096, 256, 0, stream>>>(w_out, woutb, (size_t)DFFn * Dn);

  // 2) QKV projection (bf16 out)
  k_gemm_bf16<2><<<dim3(3 * Dn / 128, Bn * Tn / 128), 256, 0, stream>>>(
      xb, wqkvb, b_qkv, nullptr, qkvb, Bn * Tn, 3 * Dn, Dn);

  // 3) repack to per-head Q/K/V
  k_repack_qkv<<<(unsigned)((size_t)Bn * Tn * 3 * Dn / 256), 256, 0, stream>>>(
      qkvb, Qb, Kb, Vb);

  // 4) flash attention
  k_attention<<<dim3(Tn / 64, Bn * Hn), 128, 0, stream>>>(Qb, Kb, Vb, mask, yattn);

  // 5) residual + LN1 (f32 + bf16 outputs)
  k_resid_ln<3><<<Bn * Tn, 256, 0, stream>>>(x, yattn, ln1w, ln1b, x1, x1b);

  // 6) FFN up + ReLU (bf16 out)
  k_gemm_bf16<6><<<dim3(DFFn / 128, Bn * Tn / 128), 256, 0, stream>>>(
      x1b, wffb, b_ff, nullptr, h1b, Bn * Tn, DFFn, Dn);

  // 7) FFN down (f32 out)
  k_gemm_bf16<1><<<dim3(Dn / 128, Bn * Tn / 128), 256, 0, stream>>>(
      h1b, woutb, b_out, y2, nullptr, Bn * Tn, Dn, DFFn);

  // 8) residual + LN2 -> final output
  k_resid_ln<1><<<Bn * Tn, 256, 0, stream>>>(x1, y2, ln2w, ln2b, out, nullptr);
}
